// CharRNN_25580825215452
// MI455X (gfx1250) — compile-verified, matrix operand order checked by reading, not run on hardware
//
#include <hip/hip_runtime.h>

#define B_ 64
#define T_ 256
#define V_ 8000
#define R_ 512

typedef __attribute__((ext_vector_type(16))) __bf16 v16bf;
typedef __attribute__((ext_vector_type(8)))  float  v8f;

// ---------------- CDNA5 helpers ----------------

#if defined(__gfx1250__) && __has_builtin(__builtin_amdgcn_global_load_async_to_lds_b128)
#define ASYNC_LDS 1
#else
#define ASYNC_LDS 0
#endif

#if ASYNC_LDS
typedef int v4i_ __attribute__((vector_size(16)));
typedef __attribute__((address_space(1))) v4i_ gv4i;   // global (AS1) int4
typedef __attribute__((address_space(3))) v4i_ lv4i;   // LDS (AS3) int4
#endif

// Copy 16 bytes global -> LDS (async on gfx1250 when the builtin exists).
static __device__ __forceinline__ void cp16(void* ldst, const void* gsrc) {
#if ASYNC_LDS
  __builtin_amdgcn_global_load_async_to_lds_b128(
      (gv4i*)(unsigned long long)gsrc,
      (lv4i*)(unsigned long long)ldst,
      0, 0);
#else
  *(uint4*)ldst = *(const uint4*)gsrc;
#endif
}

static __device__ __forceinline__ void wait_async() {
#if ASYNC_LDS
#if __has_builtin(__builtin_amdgcn_s_wait_asynccnt)
  __builtin_amdgcn_s_wait_asynccnt(0);
#else
  asm volatile("s_wait_asynccnt 0" ::: "memory");
#endif
#endif
}

// Load a 16x16x32 bf16 WMMA fragment for this lane.
// ISA layout (A 16x32 / B stored N-major): lane half = lane>>4 selects
// K runs {8*half .. 8*half+7} and {16+8*half .. 16+8*half+7} within a k-step.
// p must point at (row_base + 32*kk + 8*half), 16B aligned.
static __device__ __forceinline__ v16bf load_frag(const __bf16* p) {
  union { v16bf v; uint4 q[2]; } u;
  u.q[0] = *(const uint4*)(p);
  u.q[1] = *(const uint4*)(p + 16);
  return u.v;
}

static __device__ __forceinline__ v8f wmma_bf16(v16bf a, v16bf b, v8f c) {
#if defined(__gfx1250__)
  return __builtin_amdgcn_wmma_f32_16x16x32_bf16(false, a, false, b,
                                                 (short)0, c, false, false);
#else
  return c;  // host-pass placeholder
#endif
}

static __device__ __forceinline__ float sigmoidf(float x) {
  return 1.0f / (1.0f + __expf(-x));
}

// ---------------- setup kernels ----------------

__global__ void k_init(__bf16* hb1, __bf16* hb2, unsigned* cnt) {
  size_t i = (size_t)blockIdx.x * blockDim.x + threadIdx.x;
  if (i < (size_t)2 * B_ * R_) { hb1[i] = (__bf16)0.0f; hb2[i] = (__bf16)0.0f; }
  if (i < 64) cnt[i] = 0u;
}

// Embedding gather -> time-major bf16 xs[t][b][r]
__global__ void k_embed(const int* __restrict__ ids, const float* __restrict__ emb,
                        __bf16* __restrict__ xs) {
  const int bt = blockIdx.x;            // bt = b*T + t
  const int b = bt / T_, t = bt % T_;
  const int idx = ids[bt];
  const float* src = emb + (size_t)idx * R_;
  __bf16* dst = xs + ((size_t)t * B_ + b) * R_;
  for (int r = threadIdx.x; r < R_; r += blockDim.x) dst[r] = (__bf16)src[r];
}

// Wt[n][k] = (bf16) W[k][n]  (transpose so per-lane B fragments are contiguous in K)
__global__ void k_cvtT(const float* __restrict__ W, __bf16* __restrict__ Wt,
                       int K, int N) {
  size_t i = (size_t)blockIdx.x * blockDim.x + threadIdx.x;
  size_t total = (size_t)K * N;
  if (i >= total) return;
  size_t n = i / (size_t)K, k = i % (size_t)K;
  Wt[i] = (__bf16)W[k * (size_t)N + n];
}

// ---------------- persistent LSTM scan ----------------
// 32 blocks x 256 threads (8 wave32 waves). Block `blk` owns h columns
// [blk*16, blk*16+16) => gate columns {g*512 + blk*16 .. +16}, g = i,j,f,o.
// LDS: wlds (weights, resident all 256 steps) 128KB | alds (A=[x_t|h]) 128KB
//      glds (gate f32 exchange) 16KB | clds (c-state) 4KB  => 276KB of 320KB.
__global__ void __launch_bounds__(256, 1)
k_lstm(const __bf16* __restrict__ in_seq, __bf16* __restrict__ out_seq,
       const __bf16* __restrict__ Wt, const float* __restrict__ bias,
       __bf16* __restrict__ hb /* ping-pong [2][64][512] */,
       unsigned* __restrict__ cnt) {
  extern __shared__ char smem[];
  __bf16* wlds = (__bf16*)smem;             // [64 rows (4g x 16 cols)][1024]
  __bf16* alds = (__bf16*)(smem + 131072);  // [64][1024]  ( [x_t | h] )
  float*  glds = (float*)(smem + 262144);   // [64][64]
  float*  clds = (float*)(smem + 278528);   // [64][16]

  const int blk  = blockIdx.x;
  const int tid  = threadIdx.x;
  const int wav  = tid >> 5;
  const int lane = tid & 31;
  const int l16  = lane & 15;
  const int half = lane >> 4;

  // Preload this block's weight slice into LDS (stays resident whole scan).
  for (int j = 0; j < 4; ++j) {
    const __bf16* src = Wt + ((size_t)(j * 512 + blk * 16)) * 1024;
    __bf16* dst = wlds + (size_t)j * 16 * 1024;
    for (int i = tid; i < 2048; i += 256) cp16(dst + i * 8, src + i * 8);
  }
  for (int i = tid; i < 1024; i += 256) clds[i] = 0.0f;
  wait_async();
  __syncthreads();

  // Wave tile assignment: m-tile = wav&3, two gate n-tiles jb, jb+1.
  const int m  = wav & 3;
  const int jb = (wav >> 2) * 2;
  const __bf16* arow  = alds + ((size_t)(m * 16 + l16)) * 1024;
  const __bf16* brow0 = wlds + ((size_t)(jb * 16 + l16)) * 1024;
  const __bf16* brow1 = wlds + ((size_t)((jb + 1) * 16 + l16)) * 1024;
  const int koff = half * 8;

  for (int t = 0; t < T_; ++t) {
    // Stage A = [x_t | h_prev] into LDS (async).
    const __bf16* xt = in_seq + (size_t)t * (B_ * R_);
    const __bf16* hp = hb + (size_t)(t & 1) * (B_ * R_);
    for (int i = tid; i < 4096; i += 256) {
      const int b = i >> 6, c = (i & 63) * 8;
      cp16(alds + (size_t)b * 1024 + c,       xt + (size_t)b * 512 + c);
      cp16(alds + (size_t)b * 1024 + 512 + c, hp + (size_t)b * 512 + c);
    }
    wait_async();
    __syncthreads();

    // GEMM: gates[64 x 64cols] = A[64 x 1024] x Wslice, K in 32 steps of 32.
    // 4 accumulator chains + register double-buffered fragments: loads of
    // iteration i+1 target fresh temps while WMMAs consume iteration i's
    // registers -> no WAR on WMMA sources, no hazard v_nops.
    v8f acc0a = {}; v8f acc0b = {}; v8f acc1a = {}; v8f acc1b = {};
    v16bf a0  = load_frag(arow  + koff);
    v16bf a1  = load_frag(arow  + koff + 32);
    v16bf b00 = load_frag(brow0 + koff);
    v16bf b10 = load_frag(brow1 + koff);
    v16bf b01 = load_frag(brow0 + koff + 32);
    v16bf b11 = load_frag(brow1 + koff + 32);
#pragma unroll 3
    for (int kk = 2; kk < 32; kk += 2) {
      const int k0 = kk * 32 + koff;
      const int k1 = k0 + 32;
      v16bf na0  = load_frag(arow  + k0);
      v16bf na1  = load_frag(arow  + k1);
      v16bf nb00 = load_frag(brow0 + k0);
      v16bf nb10 = load_frag(brow1 + k0);
      v16bf nb01 = load_frag(brow0 + k1);
      v16bf nb11 = load_frag(brow1 + k1);
      acc0a = wmma_bf16(a0, b00, acc0a);
      acc1a = wmma_bf16(a0, b10, acc1a);
      acc0b = wmma_bf16(a1, b01, acc0b);
      acc1b = wmma_bf16(a1, b11, acc1b);
      a0 = na0; a1 = na1; b00 = nb00; b10 = nb10; b01 = nb01; b11 = nb11;
    }
    acc0a = wmma_bf16(a0, b00, acc0a);
    acc1a = wmma_bf16(a0, b10, acc1a);
    acc0b = wmma_bf16(a1, b01, acc0b);
    acc1b = wmma_bf16(a1, b11, acc1b);

    // D layout: element (v, lane) -> M = v + 8*half, N = l16.
#pragma unroll
    for (int v = 0; v < 8; ++v) {
      const int row = m * 16 + v + half * 8;
      glds[row * 64 + jb * 16 + l16]       = acc0a[v] + acc0b[v];
      glds[row * 64 + (jb + 1) * 16 + l16] = acc1a[v] + acc1b[v];
    }
    __syncthreads();

    // Elementwise LSTM cell update for this block's 16 h columns.
    __bf16* hw = hb + (size_t)((t + 1) & 1) * (B_ * R_);
    __bf16* ot = out_seq + (size_t)t * (B_ * R_);
    for (int e = tid; e < 1024; e += 256) {
      const int b = e >> 4, c = e & 15;
      const int gc = blk * 16 + c;
      const float gi = glds[b * 64 +      c] + bias[       gc];
      const float gj = glds[b * 64 + 16 + c] + bias[ 512 + gc];
      const float gf = glds[b * 64 + 32 + c] + bias[1024 + gc];
      const float go = glds[b * 64 + 48 + c] + bias[1536 + gc];
      const float cp = clds[e];
      const float nc = cp * sigmoidf(gf) + sigmoidf(gi) * tanhf(gj);
      const float nh = tanhf(nc) * sigmoidf(go);
      clds[e] = nc;
      const __bf16 hv = (__bf16)nh;
      hw[(size_t)b * 512 + gc] = hv;
      ot[(size_t)b * 512 + gc] = hv;
    }

    // Device-wide step barrier (all 32 blocks resident: 1 block per WGP).
    __threadfence();
    __syncthreads();
    if (tid == 0) {
      atomicAdd(cnt, 1u);
      while (atomicAdd(cnt, 0u) < 32u * (unsigned)(t + 1)) {
        __builtin_amdgcn_s_sleep(4);
      }
    }
    __syncthreads();
    __threadfence();
  }
}

// ---------------- final projection: logits = outputs @ softmax_w + b ----------------
// M = B*T = 16384 (A row r -> seq[t= r%256][b= r/256]), N = 8000, K = 512.
// Block tile 128M x 64N, wave w owns m-strip w and all 4 n-tiles (A reuse x4).
__global__ void __launch_bounds__(256)
k_logits(const __bf16* __restrict__ seq, const __bf16* __restrict__ SWt,
         const float* __restrict__ sb, float* __restrict__ out) {
  const int nb  = blockIdx.x * 64;
  const int mb  = blockIdx.y * 128;
  const int tid = threadIdx.x;
  const int wav = tid >> 5;
  const int lane = tid & 31;
  const int l16 = lane & 15, half = lane >> 4;
  const int koff = half * 8;

  const int r  = mb + wav * 16 + l16;       // this lane's A row
  const int bb = r >> 8, tt = r & 255;      // b = r / T, t = r % T
  const __bf16* arow = seq + ((size_t)tt * B_ + bb) * R_;

  const __bf16* br0 = SWt + (size_t)(nb +      l16) * R_;
  const __bf16* br1 = SWt + (size_t)(nb + 16 + l16) * R_;
  const __bf16* br2 = SWt + (size_t)(nb + 32 + l16) * R_;
  const __bf16* br3 = SWt + (size_t)(nb + 48 + l16) * R_;

  v8f acc0 = {}; v8f acc1 = {}; v8f acc2 = {}; v8f acc3 = {};
  v16bf a  = load_frag(arow + koff);
  v16bf b0 = load_frag(br0 + koff);
  v16bf b1 = load_frag(br1 + koff);
  v16bf b2 = load_frag(br2 + koff);
  v16bf b3 = load_frag(br3 + koff);
#pragma unroll 3
  for (int kk = 1; kk < 16; ++kk) {
    const int k0 = kk * 32 + koff;
    v16bf na  = load_frag(arow + k0);
    v16bf nb0 = load_frag(br0 + k0);
    v16bf nb1 = load_frag(br1 + k0);
    v16bf nb2 = load_frag(br2 + k0);
    v16bf nb3 = load_frag(br3 + k0);
    acc0 = wmma_bf16(a, b0, acc0);
    acc1 = wmma_bf16(a, b1, acc1);
    acc2 = wmma_bf16(a, b2, acc2);
    acc3 = wmma_bf16(a, b3, acc3);
    a = na; b0 = nb0; b1 = nb1; b2 = nb2; b3 = nb3;
  }
  acc0 = wmma_bf16(a, b0, acc0);
  acc1 = wmma_bf16(a, b1, acc1);
  acc2 = wmma_bf16(a, b2, acc2);
  acc3 = wmma_bf16(a, b3, acc3);

  v8f accs[4] = {acc0, acc1, acc2, acc3};
#pragma unroll
  for (int nt = 0; nt < 4; ++nt) {
    const int col = nb + nt * 16 + l16;
    const float bv = sb[col];
#pragma unroll
    for (int v = 0; v < 8; ++v) {
      const int row = mb + wav * 16 + v + half * 8;
      out[(size_t)row * V_ + col] = accs[nt][v] + bv;
    }
  }
}

// ---------------- launcher ----------------

extern "C" void kernel_launch(void* const* d_in, const int* in_sizes, int n_in,
                              void* d_out, int out_size, void* d_ws, size_t ws_size,
                              hipStream_t stream) {
  (void)in_sizes; (void)n_in; (void)out_size; (void)ws_size;
  const int*   ids = (const int*)d_in[0];
  const float* emb = (const float*)d_in[1];
  const float* W1  = (const float*)d_in[2];
  const float* b1  = (const float*)d_in[3];
  const float* W2  = (const float*)d_in[4];
  const float* b2  = (const float*)d_in[5];
  const float* sw  = (const float*)d_in[6];
  const float* sb  = (const float*)d_in[7];
  float* out = (float*)d_out;

  char* p = (char*)d_ws;
  auto carve = [&](size_t bytes) {
    char* r = p; p += (bytes + 255) & ~(size_t)255; return r;
  };
  __bf16* xs   = (__bf16*)carve((size_t)T_ * B_ * R_ * 2);        // embedded seq (bf16)
  __bf16* seq2 = (__bf16*)carve((size_t)T_ * B_ * R_ * 2);        // layer-1 h seq
  __bf16* seq3 = (__bf16*)carve((size_t)T_ * B_ * R_ * 2);        // layer-2 h seq
  __bf16* Wt1  = (__bf16*)carve((size_t)2 * R_ * 4 * R_ * 2);     // W1^T bf16 [2048][1024]
  __bf16* Wt2  = (__bf16*)carve((size_t)2 * R_ * 4 * R_ * 2);
  __bf16* SWt  = (__bf16*)carve((size_t)V_ * R_ * 2);             // softmax_w^T [8000][512]
  __bf16* hb1  = (__bf16*)carve((size_t)2 * B_ * R_ * 2);         // h ping-pong L1
  __bf16* hb2  = (__bf16*)carve((size_t)2 * B_ * R_ * 2);         // h ping-pong L2
  unsigned* cnt = (unsigned*)carve(256);

  k_init<<<256, 256, 0, stream>>>(hb1, hb2, cnt);
  k_embed<<<B_ * T_, 256, 0, stream>>>(ids, emb, xs);

  const size_t totW = (size_t)2 * R_ * 4 * R_;
  k_cvtT<<<(unsigned)((totW + 255) / 256), 256, 0, stream>>>(W1, Wt1, 2 * R_, 4 * R_);
  k_cvtT<<<(unsigned)((totW + 255) / 256), 256, 0, stream>>>(W2, Wt2, 2 * R_, 4 * R_);
  const size_t totS = (size_t)R_ * V_;
  k_cvtT<<<(unsigned)((totS + 255) / 256), 256, 0, stream>>>(sw, SWt, R_, V_);

  const size_t lds_bytes = 131072 + 131072 + 16384 + 4096;  // 276 KB of 320 KB/WGP
  k_lstm<<<32, 256, lds_bytes, stream>>>(xs,   seq2, Wt1, b1, hb1, cnt + 0);
  k_lstm<<<32, 256, lds_bytes, stream>>>(seq2, seq3, Wt2, b2, hb2, cnt + 16);

  dim3 g(V_ / 64, (B_ * T_) / 128);
  k_logits<<<g, 256, 0, stream>>>(seq3, SWt, sb, out);
}
